// VQ_VAE_58050777972976
// MI455X (gfx1250) — compile-verified, matrix operand order checked by reading, not run on hardware
//
#include <hip/hip_runtime.h>
#include <hip/hip_bf16.h>

// ---------------------------------------------------------------------------
// VQ-VAE forward for MI455X (gfx1250, wave32, WMMA).
// All heavy ops are implicit GEMMs on v_wmma_f32_16x16x32_f16.
// Activations NHWC f16; weights repacked to [oc][kh][kw][ic] f16.
// Contiguous full tiles staged with global_load_async_to_lds_b128 (ASYNCcnt).
// Transposed convs: per-output-parity 2x2-tap GEMMs.
// ---------------------------------------------------------------------------

typedef __attribute__((ext_vector_type(16))) _Float16 v16h;
typedef __attribute__((ext_vector_type(8)))  float    v8f;

#define TM 64
#define TN 64
#define TK 32

// ---- CDNA5 async LDS fill -------------------------------------------------
// global_load_async_to_lds_b128: VDST = LDS byte address, ADDR = 64-bit
// global address (GV mode, saddr = off).  Tracked by ASYNCcnt.
__device__ __forceinline__ void async_ld16(void* ldsDst, const void* gsrc) {
    asm volatile("global_load_async_to_lds_b128 %0, %1, off"
                 :: "v"((unsigned)(unsigned long)ldsDst), "v"(gsrc)
                 : "memory");
}
__device__ __forceinline__ void wait_async0() {
    asm volatile("s_wait_asynccnt 0x0" ::: "memory");
}

// Load a 16x32 f16 fragment (A or B operand) from an LDS tile stored
// row-major [row][32].  Layout per CDNA5 ISA 7.12.2 (16-bit A-matrix 16x32):
//   lanes 0-15 : rows 0-15, K = 0..7 and 16..23
//   lanes16-31 : rows 0-15, K = 8..15 and 24..31
__device__ __forceinline__ v16h ldsFrag(const _Float16* base, int rowOff, int lane) {
    const int r = rowOff + (lane & 15);
    const int g = (lane >> 4) * 8;
    const _Float16* p = base + r * TK + g;
    v16h f;
#pragma unroll
    for (int i = 0; i < 8; ++i) { f[i] = p[i]; f[i + 8] = p[16 + i]; }
    return f;
}

__device__ __forceinline__ v8f wmma_f16(v16h a, v16h b, v8f c) {
    return __builtin_amdgcn_wmma_f32_16x16x32_f16(false, a, false, b, (short)0, c,
                                                  false, false);
}

__device__ __forceinline__ v16h zero16h() {
    v16h v;
#pragma unroll
    for (int i = 0; i < 16; ++i) v[i] = (_Float16)0.f;
    return v;
}

// ---------------------------------------------------------------------------
// Generic implicit-GEMM convolution, compile-time layer geometry.
//   out[m][oc] = sum_kd A[m][kd] * W[oc][kd],  kd = (kh,kw,ic), NHWC input.
// FUSE: 0 = none, 1 = relu, 2 = add residual (resid, NHWC f16).
// ---------------------------------------------------------------------------
template <int KH, int KW, int IC, int STRIDE, int PAD, int FUSE>
__global__ void __launch_bounds__(128)
conv_gemm(const _Float16* __restrict__ in, const _Float16* __restrict__ wgt,
          const _Float16* __restrict__ resid, _Float16* __restrict__ out,
          int Nb, int IH, int IW, int OH, int OW, int OC)
{
    constexpr int Kd = KH * KW * IC;
    __shared__ _Float16 As[TM * TK];
    __shared__ _Float16 Bs[TN * TK];

    const int tid  = threadIdx.x;
    const int lane = tid & 31;
    const int wave = tid >> 5;
    const int wm   = (wave >> 1) * 32;
    const int wn   = (wave & 1) * 32;

    const int M     = Nb * OH * OW;
    const int mBase = blockIdx.x * TM;
    const int nBase = blockIdx.y * TN;

    const int mLoc  = tid >> 1;        // staging row (0..63)
    const int kHalf = (tid & 1) * 16;  // staging K-half

    const int gm = mBase + mLoc;
    int n_img = 0, oh = 0, ow = 0;
    const bool mOK = gm < M;
    if (mOK) {
        n_img = gm / (OH * OW);
        int r = gm - n_img * OH * OW;
        oh = r / OW;
        ow = r - oh * OW;
    }
    const int ihBase = oh * STRIDE - PAD;
    const int iwBase = ow * STRIDE - PAD;
    const int gn = nBase + mLoc;

    const v8f zacc = {0.f, 0.f, 0.f, 0.f, 0.f, 0.f, 0.f, 0.f};
    v8f acc[2][2];
#pragma unroll
    for (int i = 0; i < 2; ++i)
#pragma unroll
        for (int j = 0; j < 2; ++j) acc[i][j] = zacc;

    for (int k0 = 0; k0 < Kd; k0 += TK) {
        // ---- stage B (weights, K-contiguous rows) ----
        const bool bFull = (nBase + TN <= OC) && (k0 + TK <= Kd);
        if (bFull) {
            // async LDS fill: 2 x 16B chunks per thread, 256 chunks = 4KB tile
#pragma unroll
            for (int c = 0; c < 2; ++c) {
                const int chunk = tid * 2 + c;
                const int row   = chunk >> 2;
                const int koff  = (chunk & 3) * 8;
                async_ld16(&Bs[row * TK + koff],
                           wgt + (long)(nBase + row) * Kd + k0 + koff);
            }
        } else {
#pragma unroll 4
            for (int i = 0; i < 16; ++i) {
                const int kd = k0 + kHalf + i;
                _Float16 v = (_Float16)0.f;
                if (gn < OC && kd < Kd) v = wgt[(long)gn * Kd + kd];
                Bs[mLoc * TK + kHalf + i] = v;
            }
        }

        // ---- stage A (im2col gather with zero padding) ----
        if constexpr (KH == 1 && KW == 1 && (IC % 16) == 0) {
            // pure GEMM rows: async when the whole tile is in range
            if (mBase + TM <= M) {
#pragma unroll
                for (int c = 0; c < 2; ++c) {
                    const int chunk = tid * 2 + c;
                    const int row   = chunk >> 2;
                    const int koff  = (chunk & 3) * 8;
                    async_ld16(&As[row * TK + koff],
                               in + (long)(mBase + row) * IC + k0 + koff);
                }
            } else {
                v16h v = zero16h();
                if (mOK) v = *(const v16h*)(in + (long)gm * IC + k0 + kHalf);
                *(v16h*)&As[mLoc * TK + kHalf] = v;
            }
        } else if constexpr ((IC % 16) == 0) {
            // one 16-ic chunk per thread; single (kh,kw) tap per chunk
            constexpr int KWIC = KW * IC;
            const int kd  = k0 + kHalf;
            const int kh  = kd / KWIC;
            const int rem = kd - kh * KWIC;
            const int kw  = rem / IC;
            const int ic  = rem - kw * IC;
            v16h v = zero16h();
            if (mOK) {
                const int ih = ihBase + kh;
                const int iw = iwBase + kw;
                if ((unsigned)ih < (unsigned)IH && (unsigned)iw < (unsigned)IW)
                    v = *(const v16h*)(in + (((long)n_img * IH + ih) * IW + iw) * IC + ic);
            }
            *(v16h*)&As[mLoc * TK + kHalf] = v;
        } else {
            // small-IC stem: per-element gather
#pragma unroll 4
            for (int i = 0; i < 16; ++i) {
                const int kd = k0 + kHalf + i;
                _Float16 v = (_Float16)0.f;
                if (mOK && kd < Kd) {
                    const int kh  = kd / (KW * IC);
                    const int rem = kd - kh * (KW * IC);
                    const int kw  = rem / IC;
                    const int ic  = rem - kw * IC;
                    const int ih  = ihBase + kh;
                    const int iw  = iwBase + kw;
                    if (ih >= 0 && ih < IH && iw >= 0 && iw < IW)
                        v = in[(((long)n_img * IH + ih) * IW + iw) * IC + ic];
                }
                As[mLoc * TK + kHalf + i] = v;
            }
        }

        wait_async0();
        __syncthreads();

        v16h aF0 = ldsFrag(As, wm +  0, lane);
        v16h aF1 = ldsFrag(As, wm + 16, lane);
        v16h bF0 = ldsFrag(Bs, wn +  0, lane);
        v16h bF1 = ldsFrag(Bs, wn + 16, lane);

        acc[0][0] = wmma_f16(aF0, bF0, acc[0][0]);
        acc[0][1] = wmma_f16(aF0, bF1, acc[0][1]);
        acc[1][0] = wmma_f16(aF1, bF0, acc[1][0]);
        acc[1][1] = wmma_f16(aF1, bF1, acc[1][1]);
        __syncthreads();
    }

    // ---- store (C/D layout: lanes 0-15 M=r, lanes 16-31 M=r+8) ----
    const int rowM = 8 * (lane >> 4);
    const int colN = lane & 15;
#pragma unroll
    for (int i = 0; i < 2; ++i)
#pragma unroll
        for (int j = 0; j < 2; ++j) {
            const int n = nBase + wn + j * 16 + colN;
            if (n >= OC) continue;
#pragma unroll
            for (int r = 0; r < 8; ++r) {
                const int m = mBase + wm + i * 16 + rowM + r;
                if (m >= M) continue;
                float v = acc[i][j][r];
                if (FUSE == 1)      v = v > 0.f ? v : 0.f;
                else if (FUSE == 2) v += (float)resid[(long)m * OC + n];
                out[(long)m * OC + n] = (_Float16)v;
            }
        }
}

// ---------------------------------------------------------------------------
// Transposed conv (k=4, s=2, p=1) as per-parity 2x2-tap implicit GEMM.
// blockIdx.z = parity (ph = z>>1, pw = z&1).  Weight pack layout:
//   wpack[ph][pw][oc][a][b][ic],  kh = (1-ph)+2a, kw = (1-pw)+2b.
// Input taps: ih = ((oh+1)>>1) - a, iw = ((ow+1)>>1) - b.  IC = 256.
// FINAL==0: out16 NHWC f16 relu(x+bias).  FINAL==1: out32 NCHW f32
// sigmoid(x+bias).
// ---------------------------------------------------------------------------
template <int FINAL>
__global__ void __launch_bounds__(128)
deconv_gemm(const _Float16* __restrict__ in, const _Float16* __restrict__ wpack,
            const float* __restrict__ bias, _Float16* __restrict__ out16,
            float* __restrict__ out32,
            int Nb, int IH, int IW, int OH, int OW, int OC)
{
    constexpr int IC = 256;
    constexpr int Kd = 4 * IC;
    __shared__ _Float16 As[TM * TK];
    __shared__ _Float16 Bs[TN * TK];

    const int tid  = threadIdx.x;
    const int lane = tid & 31;
    const int wave = tid >> 5;
    const int wm   = (wave >> 1) * 32;
    const int wn   = (wave & 1) * 32;

    const int ph = blockIdx.z >> 1, pw = blockIdx.z & 1;
    const int OH2 = OH >> 1, OW2 = OW >> 1;
    const int M  = Nb * OH2 * OW2;
    const _Float16* wgt = wpack + (long)blockIdx.z * OC * Kd;

    const int mBase = blockIdx.x * TM;
    const int nBase = blockIdx.y * TN;
    const int mLoc  = tid >> 1;
    const int kHalf = (tid & 1) * 16;

    const int gm = mBase + mLoc;
    int n_img = 0, oh = 0, ow = 0;
    const bool mOK = gm < M;
    if (mOK) {
        n_img = gm / (OH2 * OW2);
        int r = gm - n_img * OH2 * OW2;
        oh = (r / OW2) * 2 + ph;
        ow = (r - (r / OW2) * OW2) * 2 + pw;
    }
    const int ihTop = (oh + 1) >> 1;
    const int iwTop = (ow + 1) >> 1;
    const int gn = nBase + mLoc;

    const v8f zacc = {0.f, 0.f, 0.f, 0.f, 0.f, 0.f, 0.f, 0.f};
    v8f acc[2][2];
#pragma unroll
    for (int i = 0; i < 2; ++i)
#pragma unroll
        for (int j = 0; j < 2; ++j) acc[i][j] = zacc;

    for (int k0 = 0; k0 < Kd; k0 += TK) {
        // ---- stage B ----
        if (nBase + TN <= OC) {
#pragma unroll
            for (int c = 0; c < 2; ++c) {
                const int chunk = tid * 2 + c;
                const int row   = chunk >> 2;
                const int koff  = (chunk & 3) * 8;
                async_ld16(&Bs[row * TK + koff],
                           wgt + (long)(nBase + row) * Kd + k0 + koff);
            }
        } else {
#pragma unroll 4
            for (int i = 0; i < 16; ++i) {
                const int kd = k0 + kHalf + i;
                _Float16 v = (_Float16)0.f;
                if (gn < OC) v = wgt[(long)gn * Kd + kd];
                Bs[mLoc * TK + kHalf + i] = v;
            }
        }
        // ---- stage A (2x2 taps, 16 contiguous ic per chunk) ----
        {
            const int kd  = k0 + kHalf;
            const int a   = kd / (2 * IC);
            const int rem = kd - a * (2 * IC);
            const int b   = rem / IC;
            const int ic  = rem - b * IC;
            v16h v = zero16h();
            if (mOK) {
                const int ih = ihTop - a;
                const int iw = iwTop - b;
                if ((unsigned)ih < (unsigned)IH && (unsigned)iw < (unsigned)IW)
                    v = *(const v16h*)(in + (((long)n_img * IH + ih) * IW + iw) * IC + ic);
            }
            *(v16h*)&As[mLoc * TK + kHalf] = v;
        }

        wait_async0();
        __syncthreads();

        v16h aF0 = ldsFrag(As, wm +  0, lane);
        v16h aF1 = ldsFrag(As, wm + 16, lane);
        v16h bF0 = ldsFrag(Bs, wn +  0, lane);
        v16h bF1 = ldsFrag(Bs, wn + 16, lane);

        acc[0][0] = wmma_f16(aF0, bF0, acc[0][0]);
        acc[0][1] = wmma_f16(aF0, bF1, acc[0][1]);
        acc[1][0] = wmma_f16(aF1, bF0, acc[1][0]);
        acc[1][1] = wmma_f16(aF1, bF1, acc[1][1]);
        __syncthreads();
    }

    const int rowM = 8 * (lane >> 4);
    const int colN = lane & 15;
#pragma unroll
    for (int i = 0; i < 2; ++i)
#pragma unroll
        for (int j = 0; j < 2; ++j) {
            const int n = nBase + wn + j * 16 + colN;
            if (n >= OC) continue;
#pragma unroll
            for (int r = 0; r < 8; ++r) {
                const int m = mBase + wm + i * 16 + rowM + r;
                if (m >= M) continue;
                const int ni = m / (OH2 * OW2);
                const int rr = m - ni * OH2 * OW2;
                const int so = (rr / OW2) * 2 + ph;
                const int sw = (rr - (rr / OW2) * OW2) * 2 + pw;
                float v = acc[i][j][r] + bias[n];
                if (FINAL) {
                    v = 1.f / (1.f + __expf(-v));
                    out32[(((long)ni * OC + n) * OH + so) * OW + sw] = v;
                } else {
                    v = v > 0.f ? v : 0.f;
                    out16[(((long)ni * OH + so) * OW + sw) * OC + n] = (_Float16)v;
                }
            }
        }
}

// ---------------------------------------------------------------------------
// VQ argmin:  idx[m] = argmin_k ( ||e_k||^2 - 2 * ze_m . e_k )
// WMMA GEMM (M=32768, N=K=512, Kd=D=256), fully async-staged tiles,
// running per-lane min, cross-lane reduce through LDS.
// ---------------------------------------------------------------------------
__global__ void __launch_bounds__(128)
vq_argmin(const _Float16* __restrict__ ze, const _Float16* __restrict__ emb,
          const float* __restrict__ enorm, int* __restrict__ idxOut,
          int M, int Dd, int Kk)
{
    __shared__ _Float16 As[TM * TK];
    __shared__ _Float16 Bs[TN * TK];
    __shared__ float    redV[64][32];
    __shared__ int      redI[64][32];

    const int tid  = threadIdx.x;
    const int lane = tid & 31;
    const int wave = tid >> 5;
    const int wm   = (wave >> 1) * 32;
    const int wn   = (wave & 1) * 32;
    const int rowM = 8 * (lane >> 4);
    const int colN = lane & 15;

    const int mBase = blockIdx.x * TM;

    float best[16];
    int   bestI[16];
#pragma unroll
    for (int e = 0; e < 16; ++e) { best[e] = 3.4e38f; bestI[e] = 0; }

    const v8f zacc = {0.f, 0.f, 0.f, 0.f, 0.f, 0.f, 0.f, 0.f};

    for (int n0 = 0; n0 < Kk; n0 += TN) {
        v8f acc[2][2];
#pragma unroll
        for (int i = 0; i < 2; ++i)
#pragma unroll
            for (int j = 0; j < 2; ++j) acc[i][j] = zacc;

        for (int k0 = 0; k0 < Dd; k0 += TK) {
            // M is a multiple of TM and Kk of TN: tiles are always full
#pragma unroll
            for (int c = 0; c < 2; ++c) {
                const int chunk = tid * 2 + c;
                const int row   = chunk >> 2;
                const int koff  = (chunk & 3) * 8;
                async_ld16(&As[row * TK + koff],
                           ze + (long)(mBase + row) * Dd + k0 + koff);
                async_ld16(&Bs[row * TK + koff],
                           emb + (long)(n0 + row) * Dd + k0 + koff);
            }
            wait_async0();
            __syncthreads();

            v16h aF0 = ldsFrag(As, wm +  0, lane);
            v16h aF1 = ldsFrag(As, wm + 16, lane);
            v16h bF0 = ldsFrag(Bs, wn +  0, lane);
            v16h bF1 = ldsFrag(Bs, wn + 16, lane);

            acc[0][0] = wmma_f16(aF0, bF0, acc[0][0]);
            acc[0][1] = wmma_f16(aF0, bF1, acc[0][1]);
            acc[1][0] = wmma_f16(aF1, bF0, acc[1][0]);
            acc[1][1] = wmma_f16(aF1, bF1, acc[1][1]);
            __syncthreads();
        }

        // running min:  score = ||e||^2 - 2*dot
#pragma unroll
        for (int j = 0; j < 2; ++j) {
            const int n = n0 + wn + j * 16 + colN;
            const float en = enorm[n];
#pragma unroll
            for (int i = 0; i < 2; ++i)
#pragma unroll
                for (int r = 0; r < 8; ++r) {
                    const int e = i * 8 + r;
                    const float sc = en - 2.f * acc[i][j][r];
                    if (sc < best[e] || (sc == best[e] && n < bestI[e])) {
                        best[e] = sc; bestI[e] = n;
                    }
                }
        }
    }

    // cross-lane reduce: each (m,slot) written exactly once
#pragma unroll
    for (int i = 0; i < 2; ++i)
#pragma unroll
        for (int r = 0; r < 8; ++r) {
            const int e = i * 8 + r;
            const int mLocal = wm + i * 16 + rowM + r;
            const int slot   = (wave & 1) * 16 + colN;
            redV[mLocal][slot] = best[e];
            redI[mLocal][slot] = bestI[e];
        }
    __syncthreads();

    if (tid < 64) {
        float bv = 3.4e38f; int bi = 0;
        for (int s = 0; s < 32; ++s) {
            const float v = redV[tid][s];
            const int   k = redI[tid][s];
            if (v < bv || (v == bv && k < bi)) { bv = v; bi = k; }
        }
        const int m = mBase + tid;
        if (m < M) idxOut[m] = bi;
    }
}

// ---------------------------------------------------------------------------
// Codebook gather + loss. loss1 and loss2 have identical forward values.
// ---------------------------------------------------------------------------
__global__ void vq_gather_loss(const _Float16* __restrict__ ze,
                               const float* __restrict__ embed,
                               const int* __restrict__ idx,
                               _Float16* __restrict__ zq,
                               float* __restrict__ lossOut, int M, int Dd)
{
    const int m = blockIdx.x * blockDim.x + threadIdx.x;
    if (m >= M) return;
    const int k = idx[m];
    const float* e = embed + (long)k * Dd;
    float s = 0.f;
    for (int d = 0; d < Dd; ++d) {
        const float q = e[d];
        zq[(long)m * Dd + d] = (_Float16)q;
        const float diff = (float)ze[(long)m * Dd + d] - q;
        s += diff * diff;
    }
    s /= (float)M;
    atomicAdd(lossOut, s);
    atomicAdd(lossOut + 1, s);
}

// ---------------------------------------------------------------------------
// small prep kernels
// ---------------------------------------------------------------------------
__global__ void zero_losses(float* p) { if (threadIdx.x < 2) p[threadIdx.x] = 0.f; }

// OIHW f32 -> [oc][kh][kw][ic] f16
__global__ void cvt_conv_w(const float* __restrict__ w, _Float16* __restrict__ out,
                           int OC, int IC, int KH, int KW)
{
    const int e = blockIdx.x * blockDim.x + threadIdx.x;
    const int tot = OC * IC * KH * KW;
    if (e >= tot) return;
    int t  = e;
    const int ic = t % IC; t /= IC;
    const int kw = t % KW; t /= KW;
    const int kh = t % KH; const int oc = t / KH;
    out[e] = (_Float16)w[(((long)oc * IC + ic) * KH + kh) * KW + kw];
}

// IOHW f32 -> [ph][pw][oc][a][b][ic] f16,  kh=(1-ph)+2a, kw=(1-pw)+2b
__global__ void cvt_deconv_w(const float* __restrict__ w, _Float16* __restrict__ out,
                             int IC, int OC)
{
    const int e = blockIdx.x * blockDim.x + threadIdx.x;
    const int tot = 16 * OC * IC;
    if (e >= tot) return;
    int t  = e;
    const int ic = t % IC; t /= IC;
    const int b  = t & 1;  t >>= 1;
    const int a  = t & 1;  t >>= 1;
    const int oc = t % OC; t /= OC;
    const int pw = t & 1;  const int ph = t >> 1;
    const int kh = (1 - ph) + 2 * a;
    const int kw = (1 - pw) + 2 * b;
    out[e] = (_Float16)w[(((long)ic * OC + oc) * 4 + kh) * 4 + kw];
}

// NCHW f32 -> NHWC f16
__global__ void cvt_nchw_to_nhwc(const float* __restrict__ x, _Float16* __restrict__ out,
                                 int Nb, int C, int H, int W)
{
    const int e = blockIdx.x * blockDim.x + threadIdx.x;
    const int tot = Nb * C * H * W;
    if (e >= tot) return;
    int t = e;
    const int c = t % C; t /= C;
    const int w = t % W; t /= W;
    const int h = t % H; const int n = t / H;
    out[e] = (_Float16)x[(((long)n * C + c) * H + h) * W + w];
}

// codebook f32 -> f16 copy + row norms
__global__ void embed_prep(const float* __restrict__ emb, _Float16* __restrict__ embh,
                           float* __restrict__ enorm, int Kk, int Dd)
{
    const int k = blockIdx.x * blockDim.x + threadIdx.x;
    if (k >= Kk) return;
    float s = 0.f;
    for (int d = 0; d < Dd; ++d) {
        const float v = emb[(long)k * Dd + d];
        embh[(long)k * Dd + d] = (_Float16)v;
        s += v * v;
    }
    enorm[k] = s;
}

// ---------------------------------------------------------------------------
// host-side launcher
// ---------------------------------------------------------------------------
extern "C" void kernel_launch(void* const* d_in, const int* in_sizes, int n_in,
                              void* d_out, int out_size, void* d_ws, size_t ws_size,
                              hipStream_t stream)
{
    (void)in_sizes; (void)n_in; (void)out_size; (void)ws_size;

    const int Bn = 32, CIN = 3, IMG = 128, D = 256, K = 512;
    const int H1 = 64, H2 = 32;

    const float* x       = (const float*)d_in[0];
    const float* embed_w = (const float*)d_in[1];
    const float* e_w1    = (const float*)d_in[2];
    const float* e_w2    = (const float*)d_in[3];
    const float* e_r1a   = (const float*)d_in[4];
    const float* e_r1b   = (const float*)d_in[5];
    const float* e_r2a   = (const float*)d_in[6];
    const float* e_r2b   = (const float*)d_in[7];
    const float* d_r1a   = (const float*)d_in[8];
    const float* d_r1b   = (const float*)d_in[9];
    const float* d_r2a   = (const float*)d_in[10];
    const float* d_r2b   = (const float*)d_in[11];
    const float* dt1_w   = (const float*)d_in[12];
    const float* dt1_b   = (const float*)d_in[13];
    const float* dt2_w   = (const float*)d_in[14];
    const float* dt2_b   = (const float*)d_in[15];

    char* wp = (char*)d_ws;
    auto alloc = [&](size_t bytes) -> void* {
        void* r = (void*)wp;
        wp += (bytes + 255) & ~(size_t)255;
        return r;
    };
    _Float16* e_w1h  = (_Float16*)alloc(sizeof(_Float16) * 256 * 48);
    _Float16* e_w2h  = (_Float16*)alloc(sizeof(_Float16) * 256 * 4096);
    _Float16* e_r1ah = (_Float16*)alloc(sizeof(_Float16) * 256 * 2304);
    _Float16* e_r1bh = (_Float16*)alloc(sizeof(_Float16) * 256 * 256);
    _Float16* e_r2ah = (_Float16*)alloc(sizeof(_Float16) * 256 * 2304);
    _Float16* e_r2bh = (_Float16*)alloc(sizeof(_Float16) * 256 * 256);
    _Float16* d_r1ah = (_Float16*)alloc(sizeof(_Float16) * 256 * 2304);
    _Float16* d_r1bh = (_Float16*)alloc(sizeof(_Float16) * 256 * 256);
    _Float16* d_r2ah = (_Float16*)alloc(sizeof(_Float16) * 256 * 2304);
    _Float16* d_r2bh = (_Float16*)alloc(sizeof(_Float16) * 256 * 256);
    _Float16* dt1h   = (_Float16*)alloc(sizeof(_Float16) * 16 * 256 * 256);
    _Float16* dt2h   = (_Float16*)alloc(sizeof(_Float16) * 16 * 3 * 256);
    _Float16* embh   = (_Float16*)alloc(sizeof(_Float16) * (size_t)K * D);
    float*    enorm  = (float*)alloc(sizeof(float) * K);
    int*      idx    = (int*)alloc(sizeof(int) * Bn * H2 * H2);
    _Float16* xh     = (_Float16*)alloc(sizeof(_Float16) * (size_t)Bn * IMG * IMG * CIN);
    _Float16* a1     = (_Float16*)alloc(sizeof(_Float16) * (size_t)Bn * H1 * H1 * D);
    _Float16* actA   = (_Float16*)alloc(sizeof(_Float16) * (size_t)Bn * H2 * H2 * D);
    _Float16* actB   = (_Float16*)alloc(sizeof(_Float16) * (size_t)Bn * H2 * H2 * D);
    _Float16* actT   = (_Float16*)alloc(sizeof(_Float16) * (size_t)Bn * H2 * H2 * D);
    _Float16* ze     = (_Float16*)alloc(sizeof(_Float16) * (size_t)Bn * H2 * H2 * D);
    _Float16* zq     = (_Float16*)alloc(sizeof(_Float16) * (size_t)Bn * H2 * H2 * D);

    float* recon   = (float*)d_out;
    float* lossOut = recon + (size_t)Bn * CIN * IMG * IMG;

    auto up = [](int a, int b) { return (a + b - 1) / b; };

    // ---- prep ----
    zero_losses<<<1, 32, 0, stream>>>(lossOut);
    {
        int t;
        t = Bn * CIN * IMG * IMG;
        cvt_nchw_to_nhwc<<<up(t, 256), 256, 0, stream>>>(x, xh, Bn, CIN, IMG, IMG);
        t = 256 * 48;   cvt_conv_w<<<up(t, 256), 256, 0, stream>>>(e_w1,  e_w1h,  256, CIN, 4, 4);
        t = 256 * 4096; cvt_conv_w<<<up(t, 256), 256, 0, stream>>>(e_w2,  e_w2h,  256, 256, 4, 4);
        t = 256 * 2304; cvt_conv_w<<<up(t, 256), 256, 0, stream>>>(e_r1a, e_r1ah, 256, 256, 3, 3);
        t = 256 * 256;  cvt_conv_w<<<up(t, 256), 256, 0, stream>>>(e_r1b, e_r1bh, 256, 256, 1, 1);
        t = 256 * 2304; cvt_conv_w<<<up(t, 256), 256, 0, stream>>>(e_r2a, e_r2ah, 256, 256, 3, 3);
        t = 256 * 256;  cvt_conv_w<<<up(t, 256), 256, 0, stream>>>(e_r2b, e_r2bh, 256, 256, 1, 1);
        t = 256 * 2304; cvt_conv_w<<<up(t, 256), 256, 0, stream>>>(d_r1a, d_r1ah, 256, 256, 3, 3);
        t = 256 * 256;  cvt_conv_w<<<up(t, 256), 256, 0, stream>>>(d_r1b, d_r1bh, 256, 256, 1, 1);
        t = 256 * 2304; cvt_conv_w<<<up(t, 256), 256, 0, stream>>>(d_r2a, d_r2ah, 256, 256, 3, 3);
        t = 256 * 256;  cvt_conv_w<<<up(t, 256), 256, 0, stream>>>(d_r2b, d_r2bh, 256, 256, 1, 1);
        t = 16 * 256 * 256; cvt_deconv_w<<<up(t, 256), 256, 0, stream>>>(dt1_w, dt1h, 256, 256);
        t = 16 * 3 * 256;   cvt_deconv_w<<<up(t, 256), 256, 0, stream>>>(dt2_w, dt2h, 256, 3);
        embed_prep<<<up(K, 256), 256, 0, stream>>>(embed_w, embh, enorm, K, D);
    }

    const dim3 blk(128);
    // ---- encoder ----
    {   // conv1: 3->256, 128->64, k4 s2 p1, relu
        dim3 g(up(Bn * H1 * H1, TM), up(D, TN));
        conv_gemm<4, 4, 3, 2, 1, 1><<<g, blk, 0, stream>>>(
            xh, e_w1h, (const _Float16*)nullptr, a1, Bn, IMG, IMG, H1, H1, D);
    }
    {   // conv2: 256->256, 64->32, relu
        dim3 g(up(Bn * H2 * H2, TM), up(D, TN));
        conv_gemm<4, 4, 256, 2, 1, 1><<<g, blk, 0, stream>>>(
            a1, e_w2h, (const _Float16*)nullptr, actA, Bn, H1, H1, H2, H2, D);
    }
    const dim3 g32(up(Bn * H2 * H2, TM), up(D, TN));
    // res e1
    conv_gemm<3, 3, 256, 1, 1, 1><<<g32, blk, 0, stream>>>(
        actA, e_r1ah, (const _Float16*)nullptr, actT, Bn, H2, H2, H2, H2, D);
    conv_gemm<1, 1, 256, 1, 0, 2><<<g32, blk, 0, stream>>>(
        actT, e_r1bh, actA, actB, Bn, H2, H2, H2, H2, D);
    // res e2 -> z_e
    conv_gemm<3, 3, 256, 1, 1, 1><<<g32, blk, 0, stream>>>(
        actB, e_r2ah, (const _Float16*)nullptr, actT, Bn, H2, H2, H2, H2, D);
    conv_gemm<1, 1, 256, 1, 0, 2><<<g32, blk, 0, stream>>>(
        actT, e_r2bh, actB, ze, Bn, H2, H2, H2, H2, D);

    // ---- vector quantisation ----
    const int M = Bn * H2 * H2;
    vq_argmin<<<dim3(up(M, TM)), blk, 0, stream>>>(ze, embh, enorm, idx, M, D, K);
    vq_gather_loss<<<up(M, 256), 256, 0, stream>>>(ze, embed_w, idx, zq, lossOut, M, D);

    // ---- decoder ----
    conv_gemm<3, 3, 256, 1, 1, 1><<<g32, blk, 0, stream>>>(
        zq, d_r1ah, (const _Float16*)nullptr, actT, Bn, H2, H2, H2, H2, D);
    conv_gemm<1, 1, 256, 1, 0, 2><<<g32, blk, 0, stream>>>(
        actT, d_r1bh, zq, actA, Bn, H2, H2, H2, H2, D);
    conv_gemm<3, 3, 256, 1, 1, 1><<<g32, blk, 0, stream>>>(
        actA, d_r2ah, (const _Float16*)nullptr, actT, Bn, H2, H2, H2, H2, D);
    conv_gemm<1, 1, 256, 1, 0, 2><<<g32, blk, 0, stream>>>(
        actT, d_r2bh, actA, actB, Bn, H2, H2, H2, H2, D);
    {   // deconv1: 256->256, 32->64, relu(+bias)
        dim3 g(up(Bn * H2 * H2, TM), up(D, TN), 4);
        deconv_gemm<0><<<g, blk, 0, stream>>>(actB, dt1h, dt1_b, a1, (float*)nullptr,
                                              Bn, H2, H2, H1, H1, D);
    }
    {   // deconv2: 256->3, 64->128, sigmoid(+bias), NCHW f32 output
        dim3 g(up(Bn * H1 * H1, TM), 1, 4);
        deconv_gemm<1><<<g, blk, 0, stream>>>(a1, dt2h, dt2_b, (_Float16*)nullptr, recon,
                                              Bn, H1, H1, IMG, IMG, CIN);
    }
}
